// VGG_Cifar10_69965017252752
// MI455X (gfx1250) — 8-bit matrix operand order repaired and checked by reading, compiled, not run on hardware
//
#include <hip/hip_runtime.h>
#include <stdint.h>

typedef __attribute__((ext_vector_type(8))) int v8i;
typedef __attribute__((ext_vector_type(4))) int v4i;
typedef __attribute__((ext_vector_type(2))) int v2i;

#define EPSV 1e-5f


__device__ __forceinline__ v8i load_bfrag(const int8_t* wrow, int ko, int half) {
    const v4i t0 = *(const v4i*)(wrow + ko + half * 16);
    const v4i t1 = *(const v4i*)(wrow + ko + half * 16 + 32);
    v8i b;
    b[0] = t0.x; b[1] = t0.y; b[2] = t0.z; b[3] = t0.w;
    b[4] = t1.x; b[5] = t1.y; b[6] = t1.z; b[7] = t1.w;
    return b;
}

__device__ __forceinline__ v8i load_afrag(const int8_t* arow, int ko, int half) {
    v8i a;
    #pragma unroll
    for (int p = 0; p < 4; ++p) {
        const v2i t = *(const v2i*)(arow + ko + (p >> 1) * 32 + (p & 1) * 8 + half * 16);
        a[2 * p]     = t.x;
        a[2 * p + 1] = t.y;
    }
    return a;
}

__global__ __launch_bounds__(32)
void bconv32x32(const int8_t* __restrict__ act, const int8_t* __restrict__ wt,
                const int8_t* __restrict__ zbuf,
                float* __restrict__ outf, int8_t* __restrict__ outb,
                const float* __restrict__ bias,
                const float* __restrict__ bng, const float* __restrict__ bnb,
                const float* __restrict__ bnm, const float* __restrict__ bnv,
                int H, int W, int Ci, int Co, int mode)
{
    const int lane = threadIdx.x;
    const int half = lane >> 4;
    const int l16  = lane & 15;
    const int K    = 9 * Ci;
    const int m_base  = blockIdx.x * 32;
    const int co_base = blockIdx.y * 32;

    const int mA = m_base + l16;
    const int mB = mA + 16;
    const int owA = mA % W; const int tA = mA / W; const int ohA = tA % H; const int nA = tA / H;
    const int owB = mB % W; const int tB = mB / W; const int ohB = tB % H; const int nB = tB / H;

    const int co0 = co_base + l16;
    const int co1 = co0 + 16;
    const int8_t* wrow0 = wt + (size_t)co0 * K;
    const int8_t* wrow1 = wt + (size_t)co1 * K;

    v8i acc00 = {0,0,0,0,0,0,0,0};
    v8i acc01 = {0,0,0,0,0,0,0,0};
    v8i acc10 = {0,0,0,0,0,0,0,0};
    v8i acc11 = {0,0,0,0,0,0,0,0};

    #pragma unroll 3
    for (int khw = 0; khw < 9; ++khw) {
        const int kh = khw / 3, kw = khw % 3;
        const int ihA = ohA + kh - 1, iwA = owA + kw - 1;
        const int ihB = ohB + kh - 1, iwB = owB + kw - 1;
        const bool vA = (ihA >= 0) && (ihA < H) && (iwA >= 0) && (iwA < W);
        const bool vB = (ihB >= 0) && (ihB < H) && (iwB >= 0) && (iwB < W);
        const int8_t* arowA = vA ? (act + (size_t)((nA * H + ihA) * W + iwA) * Ci) : zbuf;
        const int8_t* arowB = vB ? (act + (size_t)((nB * H + ihB) * W + iwB) * Ci) : zbuf;
        const int kg = khw * Ci;
        for (int ci0 = 0; ci0 < Ci; ci0 += 64) {
            const v8i a0 = load_afrag(arowA, ci0, half);
            const v8i a1 = load_afrag(arowB, ci0, half);
            const v8i b0 = load_bfrag(wrow0, kg + ci0, half);
            const v8i b1 = load_bfrag(wrow1, kg + ci0, half);
            acc00 = __builtin_amdgcn_wmma_i32_16x16x64_iu8(true, a0, true, b0, acc00, false, false);
            acc01 = __builtin_amdgcn_wmma_i32_16x16x64_iu8(true, a0, true, b1, acc01, false, false);
            acc10 = __builtin_amdgcn_wmma_i32_16x16x64_iu8(true, a1, true, b0, acc10, false, false);
            acc11 = __builtin_amdgcn_wmma_i32_16x16x64_iu8(true, a1, true, b1, acc11, false, false);
        }
    }

    if (mode == 0) {
        const float bb0 = bias ? bias[co0] : 0.0f;
        const float bb1 = bias ? bias[co1] : 0.0f;
        #pragma unroll
        for (int r = 0; r < 8; ++r) {
            const int mr0 = m_base + r + half * 8;
            const int mr1 = mr0 + 16;
            outf[(size_t)mr0 * Co + co0] = (float)acc00[r] + bb0;
            outf[(size_t)mr0 * Co + co1] = (float)acc01[r] + bb1;
            outf[(size_t)mr1 * Co + co0] = (float)acc10[r] + bb0;
            outf[(size_t)mr1 * Co + co1] = (float)acc11[r] + bb1;
        }
    } else {
        const float s0 = bng[co0] * rsqrtf(bnv[co0] + EPSV);
        const float t0 = bnb[co0] - bnm[co0] * s0;
        const float s1 = bng[co1] * rsqrtf(bnv[co1] + EPSV);
        const float t1 = bnb[co1] - bnm[co1] * s1;
        #pragma unroll
        for (int r = 0; r < 8; ++r) {
            const int mr0 = m_base + r + half * 8;
            const int mr1 = mr0 + 16;
            outb[(size_t)mr0 * Co + co0] = ((float)acc00[r] * s0 + t0 >= 0.0f) ? (int8_t)1 : (int8_t)-1;
            outb[(size_t)mr0 * Co + co1] = ((float)acc01[r] * s1 + t1 >= 0.0f) ? (int8_t)1 : (int8_t)-1;
            outb[(size_t)mr1 * Co + co0] = ((float)acc10[r] * s0 + t0 >= 0.0f) ? (int8_t)1 : (int8_t)-1;
            outb[(size_t)mr1 * Co + co1] = ((float)acc11[r] * s1 + t1 >= 0.0f) ? (int8_t)1 : (int8_t)-1;
        }
    }
}

__global__ __launch_bounds__(32)
void bfc16x32(const int8_t* __restrict__ act, const int8_t* __restrict__ wt,
              float* __restrict__ outf, int8_t* __restrict__ outb,
              const float* __restrict__ bng, const float* __restrict__ bnb,
              const float* __restrict__ bnm, const float* __restrict__ bnv,
              int K, int Co, int mode)
{
    const int lane = threadIdx.x;
    const int half = lane >> 4;
    const int l16  = lane & 15;
    const int m_base  = blockIdx.x * 16;
    const int co_base = blockIdx.y * 32;

    const int co0 = co_base + l16;
    const int co1 = co0 + 16;
    const bool c0ok = co0 < Co, c1ok = co1 < Co;
    const int8_t* wrow0 = wt + (size_t)(c0ok ? co0 : 0) * K;
    const int8_t* wrow1 = wt + (size_t)(c1ok ? co1 : 0) * K;
    const int8_t* arow  = act + (size_t)(m_base + l16) * K;

    v8i acc0 = {0,0,0,0,0,0,0,0};
    v8i acc1 = {0,0,0,0,0,0,0,0};

    for (int k0 = 0; k0 < K; k0 += 64) {
        const v8i a  = load_afrag(arow, k0, half);
        const v8i b0 = load_bfrag(wrow0, k0, half);
        const v8i b1 = load_bfrag(wrow1, k0, half);
        acc0 = __builtin_amdgcn_wmma_i32_16x16x64_iu8(true, a, true, b0, acc0, false, false);
        acc1 = __builtin_amdgcn_wmma_i32_16x16x64_iu8(true, a, true, b1, acc1, false, false);
    }

    if (mode == 1) {
        const float s0 = c0ok ? bng[co0] * rsqrtf(bnv[co0] + EPSV) : 0.0f;
        const float t0 = c0ok ? bnb[co0] - bnm[co0] * s0 : 0.0f;
        const float s1 = c1ok ? bng[co1] * rsqrtf(bnv[co1] + EPSV) : 0.0f;
        const float t1 = c1ok ? bnb[co1] - bnm[co1] * s1 : 0.0f;
        #pragma unroll
        for (int r = 0; r < 8; ++r) {
            const int mm = m_base + r + half * 8;
            if (c0ok) outb[(size_t)mm * Co + co0] = ((float)acc0[r] * s0 + t0 >= 0.0f) ? (int8_t)1 : (int8_t)-1;
            if (c1ok) outb[(size_t)mm * Co + co1] = ((float)acc1[r] * s1 + t1 >= 0.0f) ? (int8_t)1 : (int8_t)-1;
        }
    } else {
        const float s0 = c0ok ? rsqrtf(bnv[co0] + EPSV) : 0.0f;
        const float u0 = c0ok ? bnm[co0] : 0.0f;
        const float s1 = c1ok ? rsqrtf(bnv[co1] + EPSV) : 0.0f;
        const float u1 = c1ok ? bnm[co1] : 0.0f;
        #pragma unroll
        for (int r = 0; r < 8; ++r) {
            const int mm = m_base + r + half * 8;
            if (c0ok) outf[(size_t)mm * Co + co0] = ((float)acc0[r] - u0) * s0;
            if (c1ok) outf[(size_t)mm * Co + co1] = ((float)acc1[r] - u1) * s1;
        }
    }
}

__global__ void conv1_fused(const float* __restrict__ x, const float* __restrict__ w1,
                            const float* __restrict__ g, const float* __restrict__ b,
                            const float* __restrict__ mu, const float* __restrict__ var,
                            int8_t* __restrict__ out)
{
    const int idx = blockIdx.x * blockDim.x + threadIdx.x;
    const int co = idx & 127;
    int t = idx >> 7;
    const int ow = t & 31; t >>= 5;
    const int oh = t & 31;
    const int n  = t >> 5;
    float s = 0.0f;
    #pragma unroll
    for (int ci = 0; ci < 3; ++ci) {
        #pragma unroll
        for (int kh = 0; kh < 3; ++kh) {
            const int ih = oh + kh - 1;
            if (ih < 0 || ih >= 32) continue;
            #pragma unroll
            for (int kw = 0; kw < 3; ++kw) {
                const int iw = ow + kw - 1;
                if (iw < 0 || iw >= 32) continue;
                const float xv = x[((n * 3 + ci) * 32 + ih) * 32 + iw];
                const float wv = w1[((co * 3 + ci) * 3 + kh) * 3 + kw];
                s += (wv >= 0.0f) ? xv : -xv;
            }
        }
    }
    const float f = (s - mu[co]) * (g[co] * rsqrtf(var[co] + EPSV)) + b[co];
    out[idx] = (f >= 0.0f) ? (int8_t)1 : (int8_t)-1;
}

__global__ void pool_bn_bin(const float* __restrict__ in, int8_t* __restrict__ out,
                            const float* __restrict__ g, const float* __restrict__ b,
                            const float* __restrict__ mu, const float* __restrict__ var,
                            int H, int W, int C, int nchw_out, int total)
{
    const int idx = blockIdx.x * blockDim.x + threadIdx.x;
    if (idx >= total) return;
    const int Ho = H >> 1, Wo = W >> 1;
    const int c = idx % C;
    int t = idx / C;
    const int xo = t % Wo; t /= Wo;
    const int yo = t % Ho;
    const int n  = t / Ho;
    const float* p = in + ((size_t)((n * H + yo * 2) * W + xo * 2)) * C + c;
    const size_t rs = (size_t)W * C;
    const float mx = fmaxf(fmaxf(p[0], p[C]), fmaxf(p[rs], p[rs + C]));
    const float f = (mx - mu[c]) * (g[c] * rsqrtf(var[c] + EPSV)) + b[c];
    const int8_t s = (f >= 0.0f) ? (int8_t)1 : (int8_t)-1;
    if (nchw_out) out[((size_t)(n * C + c) * Ho + yo) * Wo + xo] = s;
    else          out[idx] = s;
}

__global__ void repack_conv_w(const float* __restrict__ w, int8_t* __restrict__ o,
                              int Ci, int total)
{
    const int idx = blockIdx.x * blockDim.x + threadIdx.x;
    if (idx >= total) return;
    const int kw = idx % 3;
    int t = idx / 3;
    const int kh = t % 3; t /= 3;
    const int ci = t % Ci;
    const int co = t / Ci;
    const int k  = (kh * 3 + kw) * Ci + ci;
    o[(size_t)co * (9 * Ci) + k] = (w[idx] >= 0.0f) ? (int8_t)1 : (int8_t)-1;
}

__global__ void bin_w(const float* __restrict__ w, int8_t* __restrict__ o, int total)
{
    const int idx = blockIdx.x * blockDim.x + threadIdx.x;
    if (idx >= total) return;
    o[idx] = (w[idx] >= 0.0f) ? (int8_t)1 : (int8_t)-1;
}

__global__ void zero_init(int8_t* __restrict__ p, int total)
{
    const int idx = blockIdx.x * blockDim.x + threadIdx.x;
    if (idx < total) p[idx] = 0;
}

__global__ void logsoftmax10(const float* __restrict__ in, float* __restrict__ out)
{
    const int r = blockIdx.x * blockDim.x + threadIdx.x;
    if (r >= 256) return;
    const float* p = in + r * 10;
    float mx = p[0];
    #pragma unroll
    for (int i = 1; i < 10; ++i) mx = fmaxf(mx, p[i]);
    float s = 0.0f;
    #pragma unroll
    for (int i = 0; i < 10; ++i) s += __expf(p[i] - mx);
    const float l = __logf(s);
    #pragma unroll
    for (int i = 0; i < 10; ++i) out[r * 10 + i] = p[i] - mx - l;
}

extern "C" void kernel_launch(void* const* d_in, const int* in_sizes, int n_in,
                              void* d_out, int out_size, void* d_ws, size_t ws_size,
                              hipStream_t stream)
{
    (void)in_sizes; (void)n_in; (void)out_size; (void)ws_size;
    const float* x     = (const float*)d_in[0];
    const float* w1    = (const float*)d_in[1];
    const float* g1    = (const float*)d_in[2];
    const float* b1    = (const float*)d_in[3];
    const float* m1    = (const float*)d_in[4];
    const float* v1    = (const float*)d_in[5];
    const float* w2    = (const float*)d_in[6];
    const float* g2    = (const float*)d_in[7];
    const float* b2    = (const float*)d_in[8];
    const float* m2    = (const float*)d_in[9];
    const float* v2    = (const float*)d_in[10];
    const float* w3    = (const float*)d_in[11];
    const float* g3    = (const float*)d_in[12];
    const float* b3    = (const float*)d_in[13];
    const float* m3    = (const float*)d_in[14];
    const float* v3    = (const float*)d_in[15];
    const float* w4    = (const float*)d_in[16];
    const float* g4    = (const float*)d_in[17];
    const float* b4    = (const float*)d_in[18];
    const float* m4    = (const float*)d_in[19];
    const float* v4    = (const float*)d_in[20];
    const float* w5    = (const float*)d_in[21];
    const float* g5    = (const float*)d_in[22];
    const float* b5    = (const float*)d_in[23];
    const float* m5    = (const float*)d_in[24];
    const float* v5    = (const float*)d_in[25];
    const float* w6    = (const float*)d_in[26];
    const float* g6    = (const float*)d_in[27];
    const float* b6    = (const float*)d_in[28];
    const float* m6    = (const float*)d_in[29];
    const float* v6    = (const float*)d_in[30];
    const float* bias2 = (const float*)d_in[31];
    const float* w7    = (const float*)d_in[32];
    const float* m7    = (const float*)d_in[33];
    const float* v7    = (const float*)d_in[34];
    const float* g7    = (const float*)d_in[35];
    const float* b7    = (const float*)d_in[36];
    const float* w8    = (const float*)d_in[37];
    const float* m8    = (const float*)d_in[38];
    const float* v8    = (const float*)d_in[39];
    const float* g8    = (const float*)d_in[40];
    const float* b8    = (const float*)d_in[41];
    const float* w9    = (const float*)d_in[42];
    const float* m9    = (const float*)d_in[43];
    const float* v9    = (const float*)d_in[44];

    char* ws = (char*)d_ws;
    size_t off = 0;
    auto alloc = [&](size_t bytes) { size_t o = off; off += (bytes + 255) & ~(size_t)255; return o; };
    int8_t* zbuf = (int8_t*)(ws + alloc(1024));
    int8_t* wt2 = (int8_t*)(ws + alloc((size_t)128 * 1152));
    int8_t* wt3 = (int8_t*)(ws + alloc((size_t)256 * 1152));
    int8_t* wt4 = (int8_t*)(ws + alloc((size_t)256 * 2304));
    int8_t* wt5 = (int8_t*)(ws + alloc((size_t)512 * 2304));
    int8_t* wt6 = (int8_t*)(ws + alloc((size_t)512 * 4608));
    int8_t* wt7 = (int8_t*)(ws + alloc((size_t)1024 * 8192));
    int8_t* wt8 = (int8_t*)(ws + alloc((size_t)1024 * 1024));
    int8_t* wt9 = (int8_t*)(ws + alloc((size_t)10 * 1024));
    float*  fcf = (float*) (ws + alloc((size_t)256 * 10 * 4));
    float*  cvf = (float*) (ws + alloc((size_t)256 * 32 * 32 * 128 * 4));
    int8_t* bA  = (int8_t*)(ws + alloc((size_t)256 * 32 * 32 * 128));
    int8_t* bB  = (int8_t*)(ws + alloc((size_t)256 * 32 * 32 * 128));

    zero_init<<<1, 1024, 0, stream>>>(zbuf, 1024);
    repack_conv_w<<<(128 * 128 * 9 + 255) / 256, 256, 0, stream>>>(w2, wt2, 128, 128 * 128 * 9);
    repack_conv_w<<<(256 * 128 * 9 + 255) / 256, 256, 0, stream>>>(w3, wt3, 128, 256 * 128 * 9);
    repack_conv_w<<<(256 * 256 * 9 + 255) / 256, 256, 0, stream>>>(w4, wt4, 256, 256 * 256 * 9);
    repack_conv_w<<<(512 * 256 * 9 + 255) / 256, 256, 0, stream>>>(w5, wt5, 256, 512 * 256 * 9);
    repack_conv_w<<<(512 * 512 * 9 + 255) / 256, 256, 0, stream>>>(w6, wt6, 512, 512 * 512 * 9);
    bin_w<<<(1024 * 8192 + 255) / 256, 256, 0, stream>>>(w7, wt7, 1024 * 8192);
    bin_w<<<(1024 * 1024 + 255) / 256, 256, 0, stream>>>(w8, wt8, 1024 * 1024);
    bin_w<<<(10 * 1024 + 255) / 256, 256, 0, stream>>>(w9, wt9, 10 * 1024);

    conv1_fused<<<(256 * 32 * 32 * 128) / 256, 256, 0, stream>>>(x, w1, g1, b1, m1, v1, bA);

    bconv32x32<<<dim3(262144 / 32, 128 / 32), 32, 0, stream>>>(
        bA, wt2, zbuf, cvf, nullptr, bias2, nullptr, nullptr, nullptr, nullptr,
        32, 32, 128, 128, 0);
    pool_bn_bin<<<(256 * 16 * 16 * 128 + 255) / 256, 256, 0, stream>>>(
        cvf, bB, g2, b2, m2, v2, 32, 32, 128, 0, 256 * 16 * 16 * 128);

    bconv32x32<<<dim3(65536 / 32, 256 / 32), 32, 0, stream>>>(
        bB, wt3, zbuf, nullptr, bA, nullptr, g3, b3, m3, v3,
        16, 16, 128, 256, 1);

    bconv32x32<<<dim3(65536 / 32, 256 / 32), 32, 0, stream>>>(
        bA, wt4, zbuf, cvf, nullptr, nullptr, nullptr, nullptr, nullptr, nullptr,
        16, 16, 256, 256, 0);
    pool_bn_bin<<<(256 * 8 * 8 * 256 + 255) / 256, 256, 0, stream>>>(
        cvf, bB, g4, b4, m4, v4, 16, 16, 256, 0, 256 * 8 * 8 * 256);

    bconv32x32<<<dim3(16384 / 32, 512 / 32), 32, 0, stream>>>(
        bB, wt5, zbuf, nullptr, bA, nullptr, g5, b5, m5, v5,
        8, 8, 256, 512, 1);

    bconv32x32<<<dim3(16384 / 32, 512 / 32), 32, 0, stream>>>(
        bA, wt6, zbuf, cvf, nullptr, nullptr, nullptr, nullptr, nullptr, nullptr,
        8, 8, 512, 512, 0);
    pool_bn_bin<<<(256 * 4 * 4 * 512 + 255) / 256, 256, 0, stream>>>(
        cvf, bB, g6, b6, m6, v6, 8, 8, 512, 1, 256 * 4 * 4 * 512);

    bfc16x32<<<dim3(256 / 16, 1024 / 32), 32, 0, stream>>>(
        bB, wt7, nullptr, bA, g7, b7, m7, v7, 8192, 1024, 1);
    bfc16x32<<<dim3(256 / 16, 1024 / 32), 32, 0, stream>>>(
        bA, wt8, nullptr, bB, g8, b8, m8, v8, 1024, 1024, 1);
    bfc16x32<<<dim3(256 / 16, 1), 32, 0, stream>>>(
        bB, wt9, fcf, nullptr, nullptr, nullptr, m9, v9, 1024, 10, 2);

    logsoftmax10<<<1, 256, 0, stream>>>(fcf, (float*)d_out);
}
